// GraphTransformerModel_11450382811291
// MI455X (gfx1250) — matrix operand order repaired and checked by reading, not compiled, not run on hardware
//
#include <hip/hip_runtime.h>
#include <hip/hip_bf16.h>
#include <math.h>

// ---------------------------------------------------------------------------
// Graph transformer on MI455X (gfx1250, wave32).
//   - Linear layers: v_wmma_f32_16x16x32_f16, 4 WMMAs/wave/K-step.
//   - Tile staging: double-buffered global_load_async_to_lds_b128; next tile's
//     DMA overlaps current tile's WMMAs (s_wait_asynccnt 0x3 drains only the
//     older group - 3 async issues per wave per tile, in-order completion).
//   - Activations converted to f16 once (LN writes f16); weights pre-transposed
//     to f16 Wt[M][K] once per launch.
//   - Edge softmax aggregation: order-preserving uint atomicMax segment-max,
//     exp/segment-sum, scatter atomicAdd of ex*v, 1/s folded into finalize.
// ---------------------------------------------------------------------------

typedef _Float16 v16h __attribute__((ext_vector_type(16)));
typedef _Float16 h8   __attribute__((ext_vector_type(8)));
typedef _Float16 h4   __attribute__((ext_vector_type(4)));
typedef float    v8f  __attribute__((ext_vector_type(8)));

#define HEADS 4
#define HDIM  64
#define CDIM  256
#define LN_EPS 1e-5f

// ---- gfx1250 async copy global -> LDS (16B per lane), ASYNCcnt tracked ----
__device__ __forceinline__ void async_ld16(unsigned lds_addr,
                                           unsigned long long gaddr) {
  asm volatile("global_load_async_to_lds_b128 %0, %1, off"
               :: "v"(lds_addr), "v"(gaddr) : "memory");
}

// ---------------------------------------------------------------------------
// GEMM: Y[N x M](f32) = Xh[N x K](f16) @ Wt[M x K](f16)^T + bias.
// Block 256 threads = 8 waves (4 row-groups x 2 col-groups); block tile
// 128 rows x 64 cols; each wave computes 32x32 via 4 WMMA tiles per K-step.
// Double-buffered LDS staging via async copies. K multiple of 32.
// Xh padded to 128-row multiple, Wt padded to 64-col multiple.
// ---------------------------------------------------------------------------
__global__ void __launch_bounds__(256) gemm_wmma_kernel(
    const _Float16* __restrict__ Xh, const _Float16* __restrict__ Wt,
    const float* __restrict__ bias, float* __restrict__ Y,
    int N, int K, int M) {
  __shared__ __align__(16) _Float16 Xs[2][128 * 32];  // ping-pong [row][k] 16KB
  __shared__ __align__(16) _Float16 Ws[2][64 * 32];   // ping-pong [col][k] 8KB

  const int tid     = threadIdx.x;
  const int lane    = tid & 31;
  const int wave    = tid >> 5;
  const bool lo     = lane < 16;
  const int waveRow = wave >> 1;   // 0..3
  const int waveCol = wave & 1;    // 0..1
  const int rowBase = blockIdx.x * 128;
  const int colBase = blockIdx.y * 64;

  // LDS byte offsets of the two buffers (low 32 bits of flat shared address).
  const unsigned xsb0 = (unsigned)(size_t)&Xs[0][0];
  const unsigned xsb1 = (unsigned)(size_t)&Xs[1][0];
  const unsigned wsb0 = (unsigned)(size_t)&Ws[0][0];
  const unsigned wsb1 = (unsigned)(size_t)&Ws[1][0];

  // Per-thread 16B chunk coordinates (loop-invariant).
  const int xr0 = tid >> 2;                 // X chunk 0: row 0..63
  const int xr1 = (tid + 256) >> 2;         // X chunk 1: row 64..127
  const int xko = (tid & 3) * 8;            // k offset in halfs
  const int wr  = tid >> 2;                 // W chunk: col 0..63
  const _Float16* xg0 = Xh + (size_t)(rowBase + xr0) * K + xko;
  const _Float16* xg1 = Xh + (size_t)(rowBase + xr1) * K + xko;
  const _Float16* wg  = Wt + (size_t)(colBase + wr) * K + xko;
  const unsigned xo0 = (unsigned)((xr0 * 32 + xko) * 2);
  const unsigned xo1 = (unsigned)((xr1 * 32 + xko) * 2);
  const unsigned wo  = (unsigned)((wr * 32 + xko) * 2);

  v8f c0 = {0.f,0.f,0.f,0.f,0.f,0.f,0.f,0.f};
  v8f c1 = c0, c2 = c0, c3 = c0;

  // Prologue: stage tile 0 into buffer 0 (3 async issues per thread).
  async_ld16(xsb0 + xo0, (unsigned long long)(size_t)(xg0 + 0));
  async_ld16(xsb0 + xo1, (unsigned long long)(size_t)(xg1 + 0));
  async_ld16(wsb0 + wo,  (unsigned long long)(size_t)(wg + 0));

  int p = 0;
  for (int kb = 0; kb < K; kb += 32, p ^= 1) {
    const bool hasNext = (kb + 32) < K;
    if (hasNext) {
      // Stage tile t+1 into the other buffer; it stays in flight during the
      // WMMAs below (only the older group is drained by the wait).
      const unsigned xb = p ? xsb0 : xsb1;
      const unsigned wb = p ? wsb0 : wsb1;
      const int kn = kb + 32;
      async_ld16(xb + xo0, (unsigned long long)(size_t)(xg0 + kn));
      async_ld16(xb + xo1, (unsigned long long)(size_t)(xg1 + kn));
      async_ld16(wb + wo,  (unsigned long long)(size_t)(wg + kn));
      asm volatile("s_wait_asynccnt 0x3" ::: "memory");  // tile t complete
    } else {
      asm volatile("s_wait_asynccnt 0x0" ::: "memory");
    }
    __syncthreads();   // all waves' copies of tile t visible in LDS

    // A fragments (16x32 f16, wave32 ISA layout): lanes 0-15 -> K {0..7,16..23},
    // lanes 16-31 -> K {8..15,24..31}; M = lane&15 within 16-row group.
    const _Float16* xsp = &Xs[p][0];
    const _Float16* wsp = &Ws[p][0];
    const int ka0 = lo ? 0 : 8;
    const int kb0 = lo ? 0 : 8;
    v16h a[2], b[2];
#pragma unroll
    for (int rt = 0; rt < 2; ++rt) {
      const int ar = waveRow * 32 + rt * 16 + (lane & 15);
      h8 x0 = *(const h8*)&xsp[ar * 32 + ka0];
      h8 x1 = *(const h8*)&xsp[ar * 32 + ka0 + 16];
#pragma unroll
      for (int i = 0; i < 8; ++i) { a[rt][i] = x0[i]; a[rt][i + 8] = x1[i]; }
    }
#pragma unroll
    for (int ct = 0; ct < 2; ++ct) {
      const int bc = waveCol * 32 + ct * 16 + (lane & 15);
      h8 w0 = *(const h8*)&wsp[bc * 32 + kb0];
      h8 w1 = *(const h8*)&wsp[bc * 32 + kb0 + 16];
#pragma unroll
      for (int i = 0; i < 8; ++i) { b[ct][i] = w0[i]; b[ct][i + 8] = w1[i]; }
    }
    c0 = __builtin_amdgcn_wmma_f32_16x16x32_f16(false, a[0], false, b[0], (short)0, c0, false, false);
    c1 = __builtin_amdgcn_wmma_f32_16x16x32_f16(false, a[0], false, b[1], (short)0, c1, false, false);
    c2 = __builtin_amdgcn_wmma_f32_16x16x32_f16(false, a[1], false, b[0], (short)0, c2, false, false);
    c3 = __builtin_amdgcn_wmma_f32_16x16x32_f16(false, a[1], false, b[1], (short)0, c3, false, false);
    __syncthreads();   // reads of buffer p done -> next iter may overwrite it
  }

  // Epilogue. C/D layout: VGPR r, lanes 0-15 -> M=r, lanes 16-31 -> M=r+8.
  const v8f cc[4] = {c0, c1, c2, c3};
#pragma unroll
  for (int rt = 0; rt < 2; ++rt) {
#pragma unroll
    for (int ct = 0; ct < 2; ++ct) {
      const int col = colBase + waveCol * 32 + ct * 16 + (lane & 15);
      if (col < M) {
        const float bv = bias ? bias[col] : 0.f;
        const v8f cv = cc[rt * 2 + ct];
#pragma unroll
        for (int r = 0; r < 8; ++r) {
          const int row = rowBase + waveRow * 32 + rt * 16 + r + (lo ? 0 : 8);
          if (row < N) Y[(size_t)row * M + col] = cv[r] + bv;
        }
      }
    }
  }
}

// ---------------------------------------------------------------------------
// f32 -> f16 elementwise convert (n multiple of 4).
// ---------------------------------------------------------------------------
__global__ void cvt_f16_kernel(const float* __restrict__ src,
                               _Float16* __restrict__ dst, long long n) {
  const long long i = ((long long)blockIdx.x * blockDim.x + threadIdx.x) * 4;
  if (i >= n) return;
  float4 v = *(const float4*)(src + i);
  h4 o = {(_Float16)v.x, (_Float16)v.y, (_Float16)v.z, (_Float16)v.w};
  *(h4*)(dst + i) = o;
}

// Weight prep: W[K x M](f32) -> Wt[Mpad x K](f16), zero pad cols >= M.
__global__ void wt_kernel(const float* __restrict__ W, _Float16* __restrict__ Wt,
                          int K, int M, int Mpad) {
  const int idx = blockIdx.x * blockDim.x + threadIdx.x;
  if (idx >= Mpad * K) return;
  const int m = idx / K, k = idx - m * K;
  Wt[idx] = (_Float16)((m < M) ? W[(size_t)k * M + m] : 0.f);
}

// ---------------------------------------------------------------------------
// Row LayerNorm (+optional residual add) + ReLU, f32 in -> f16 out.
// blockDim == D (64 or 256).
// ---------------------------------------------------------------------------
__global__ void ln_relu_kernel(const float* __restrict__ in,
                               const float* __restrict__ add,
                               const float* __restrict__ g,
                               const float* __restrict__ b,
                               _Float16* __restrict__ out, int D) {
  extern __shared__ float red[];
  const int row = blockIdx.x;
  const int t   = threadIdx.x;
  float v = in[(size_t)row * D + t];
  if (add) v += add[(size_t)row * D + t];
  red[t] = v;
  __syncthreads();
  for (int s = D >> 1; s > 0; s >>= 1) { if (t < s) red[t] += red[t + s]; __syncthreads(); }
  const float mean = red[0] / (float)D;
  __syncthreads();
  const float d0 = v - mean;
  red[t] = d0 * d0;
  __syncthreads();
  for (int s = D >> 1; s > 0; s >>= 1) { if (t < s) red[t] += red[t + s]; __syncthreads(); }
  const float var = red[0] / (float)D;
  const float r = d0 * rsqrtf(var + LN_EPS) * g[t] + b[t];
  out[(size_t)row * D + t] = (_Float16)fmaxf(r, 0.f);
}

// ---------------------------------------------------------------------------
// Edge softmax-aggregation.
// ---------------------------------------------------------------------------
__device__ __forceinline__ unsigned enc_f(float f) {
  unsigned u = __float_as_uint(f);
  return (u & 0x80000000u) ? ~u : (u | 0x80000000u);
}
__device__ __forceinline__ float dec_f(unsigned u) {
  return (u & 0x80000000u) ? __uint_as_float(u & 0x7fffffffu)
                           : __uint_as_float(~u);
}

__global__ void edge_logits_kernel(const long long* __restrict__ ei,
                                   const float* __restrict__ q,
                                   const float* __restrict__ k,
                                   float* __restrict__ lg,
                                   unsigned* __restrict__ m, int E) {
  const long long t = (long long)blockIdx.x * blockDim.x + threadIdx.x;
  if (t >= (long long)E * HEADS) return;
  const int e = (int)(t >> 2), h = (int)(t & 3);
  const int src = (int)ei[e];
  const int dst = (int)ei[(size_t)E + e];
  const float4* qr = (const float4*)(q + (size_t)dst * CDIM + h * HDIM);
  const float4* kr = (const float4*)(k + (size_t)src * CDIM + h * HDIM);
  float s = 0.f;
#pragma unroll
  for (int i = 0; i < HDIM / 4; ++i) {
    float4 a = qr[i], b = kr[i];
    s += a.x * b.x + a.y * b.y + a.z * b.z + a.w * b.w;
  }
  s *= 0.125f;  // 1/sqrt(64)
  lg[t] = s;
  atomicMax(&m[(size_t)dst * HEADS + h], enc_f(s));
}

__global__ void edge_expsum_kernel(const long long* __restrict__ ei,
                                   float* __restrict__ lg,
                                   const unsigned* __restrict__ m,
                                   float* __restrict__ ssum, int E) {
  const long long t = (long long)blockIdx.x * blockDim.x + threadIdx.x;
  if (t >= (long long)E * HEADS) return;
  const int e = (int)(t >> 2), h = (int)(t & 3);
  const int dst = (int)ei[(size_t)E + e];
  const float ex = __expf(lg[t] - dec_f(m[(size_t)dst * HEADS + h]));
  lg[t] = ex;
  atomicAdd(&ssum[(size_t)dst * HEADS + h], ex);
}

__global__ void edge_scatter_kernel(const long long* __restrict__ ei,
                                    const float* __restrict__ lg,
                                    const float* __restrict__ v,
                                    float* __restrict__ acc, int E) {
  __shared__ int s_src, s_dst;
  __shared__ float s_ex[HEADS];
  const int e = blockIdx.x;
  const int t = threadIdx.x;
  if (t == 0) { s_src = (int)ei[e]; s_dst = (int)ei[(size_t)E + e]; }
  if (t < HEADS) s_ex[t] = lg[(size_t)e * HEADS + t];
  __syncthreads();
  const float val = s_ex[t >> 6] * v[(size_t)s_src * CDIM + t];
  atomicAdd(&acc[(size_t)s_dst * CDIM + t], val);
}

__global__ void attn_finalize_kernel(float* __restrict__ acc,
                                     const float* __restrict__ ssum,
                                     const float* __restrict__ sk, int Ncnt) {
  const size_t idx = (size_t)blockIdx.x * blockDim.x + threadIdx.x;
  if (idx >= (size_t)Ncnt * CDIM) return;
  const size_t n = idx >> 8;
  const int cc = (int)(idx & 255);
  acc[idx] = acc[idx] / (ssum[n * HEADS + (cc >> 6)] + 1e-16f) + sk[idx];
}

// ---------------------------------------------------------------------------
// Host orchestration.
// ---------------------------------------------------------------------------
extern "C" void kernel_launch(void* const* d_in, const int* in_sizes, int n_in,
                              void* d_out, int out_size, void* d_ws, size_t ws_size,
                              hipStream_t stream) {
  (void)out_size; (void)ws_size;
  const float* x      = (const float*)d_in[0];
  const float* inpW   = (const float*)d_in[1];
  const float* inpB   = (const float*)d_in[2];
  const float* lninG  = (const float*)d_in[3];
  const float* lninB  = (const float*)d_in[4];
  const float* c1W[4] = {(const float*)d_in[5], (const float*)d_in[7],
                         (const float*)d_in[9], (const float*)d_in[11]};
  const float* c1B[4] = {(const float*)d_in[6], (const float*)d_in[8],
                         (const float*)d_in[10], (const float*)d_in[12]};
  const float* res1W  = (const float*)d_in[13];
  const float* res1B  = (const float*)d_in[14];
  const float* ln1G   = (const float*)d_in[15];
  const float* ln1B   = (const float*)d_in[16];
  const float* c2W[4] = {(const float*)d_in[17], (const float*)d_in[19],
                         (const float*)d_in[21], (const float*)d_in[23]};
  const float* c2B[4] = {(const float*)d_in[18], (const float*)d_in[20],
                         (const float*)d_in[22], (const float*)d_in[24]};
  const float* res2W  = (const float*)d_in[25];
  const float* res2B  = (const float*)d_in[26];
  const float* ln2G   = (const float*)d_in[27];
  const float* ln2B   = (const float*)d_in[28];
  const float* out1W  = (const float*)d_in[29];
  const float* out1B  = (const float*)d_in[30];
  const float* lnoG   = (const float*)d_in[31];
  const float* lnoB   = (const float*)d_in[32];
  const float* out2W  = (const float*)d_in[33];
  const float* out2B  = (const float*)d_in[34];
  const long long* ei = (const long long*)d_in[n_in - 1];  // edge_index is last

  const int N    = in_sizes[0] / 64;
  const int E    = in_sizes[n_in - 1] / 2;
  const int Npad = (N + 127) & ~127;

  // ---- workspace carve-up ----
  float* ws = (float*)d_ws;
  size_t off = 0;
  auto carve = [&](size_t nfloats) { float* p = ws + off; off += nfloats; return p; };
  float* pre  = carve((size_t)N * 64);      // gemm out before 64-dim LN
  float* q    = carve((size_t)N * CDIM);
  float* kk   = carve((size_t)N * CDIM);
  float* vv   = carve((size_t)N * CDIM);
  float* sk   = carve((size_t)N * CDIM);
  float* idn  = carve((size_t)N * CDIM);
  float* acc  = carve((size_t)N * CDIM);
  float* mrow = carve((size_t)N * HEADS);   // used as unsigned
  float* ssum = carve((size_t)N * HEADS);
  float* lg   = carve((size_t)E * HEADS);

  _Float16* hbase = (_Float16*)(ws + off);
  size_t hoff = 0;
  auto carveh = [&](size_t nh) { _Float16* p = hbase + hoff; hoff += nh; return p; };
  _Float16* xh    = carveh((size_t)Npad * 64);
  _Float16* h64h  = carveh((size_t)Npad * 64);
  _Float16* h256h = carveh((size_t)Npad * CDIM);
  _Float16* wtInp  = carveh(64 * 64);
  _Float16* wtC1[4], *wtC2[4];
  for (int i = 0; i < 4; ++i) wtC1[i] = carveh((size_t)CDIM * 64);
  _Float16* wtRes1 = carveh((size_t)CDIM * 64);
  for (int i = 0; i < 4; ++i) wtC2[i] = carveh((size_t)CDIM * CDIM);
  _Float16* wtRes2 = carveh((size_t)CDIM * CDIM);
  _Float16* wtOut1 = carveh((size_t)64 * CDIM);
  _Float16* wtOut2 = carveh((size_t)64 * 64);   // M=32 padded to 64

  // ---- weight prep (once per launch) ----
  auto prepW = [&](const float* W, _Float16* Wt, int K, int M, int Mpad) {
    const int tot = Mpad * K;
    wt_kernel<<<(tot + 255) / 256, 256, 0, stream>>>(W, Wt, K, M, Mpad);
  };
  prepW(inpW, wtInp, 64, 64, 64);
  for (int i = 0; i < 4; ++i) prepW(c1W[i], wtC1[i], 64, CDIM, CDIM);
  prepW(res1W, wtRes1, 64, CDIM, CDIM);
  for (int i = 0; i < 4; ++i) prepW(c2W[i], wtC2[i], CDIM, CDIM, CDIM);
  prepW(res2W, wtRes2, CDIM, CDIM, CDIM);
  prepW(out1W, wtOut1, CDIM, 64, 64);
  prepW(out2W, wtOut2, 64, 32, 64);

  // x -> f16
  {
    const long long n = (long long)N * 64;
    cvt_f16_kernel<<<(int)((n / 4 + 255) / 256), 256, 0, stream>>>(x, xh, n);
  }

  auto gemm = [&](const _Float16* Xh, const _Float16* Wt, const float* B,
                  float* Y, int k, int m) {
    dim3 grid((N + 127) / 128, (m + 63) / 64);
    gemm_wmma_kernel<<<grid, 256, 0, stream>>>(Xh, Wt, B, Y, N, k, m);
  };
  auto lnrelu = [&](const float* in, const float* add, const float* g,
                    const float* b, _Float16* out, int d) {
    ln_relu_kernel<<<N, d, d * sizeof(float), stream>>>(in, add, g, b, out, d);
  };
  auto attn_layer = [&](const _Float16* hh, int k,
                        _Float16* const* Wq, const float* const* Bq,
                        const _Float16* resW, const float* resB,
                        const float* lnG, const float* lnB, _Float16* outh) {
    gemm(hh, Wq[0], Bq[0], q, k, CDIM);
    gemm(hh, Wq[1], Bq[1], kk, k, CDIM);
    gemm(hh, Wq[2], Bq[2], vv, k, CDIM);
    gemm(hh, Wq[3], Bq[3], sk, k, CDIM);
    gemm(hh, resW, resB, idn, k, CDIM);
    hipMemsetAsync(mrow, 0, (size_t)N * HEADS * sizeof(float), stream); // enc(-inf)
    hipMemsetAsync(ssum, 0, (size_t)N * HEADS * sizeof(float), stream);
    hipMemsetAsync(acc, 0, (size_t)N * CDIM * sizeof(float), stream);
    const int ehb = (int)(((long long)E * HEADS + 255) / 256);
    edge_logits_kernel<<<ehb, 256, 0, stream>>>(ei, q, kk, lg, (unsigned*)mrow, E);
    edge_expsum_kernel<<<ehb, 256, 0, stream>>>(ei, lg, (const unsigned*)mrow, ssum, E);
    edge_scatter_kernel<<<E, CDIM, 0, stream>>>(ei, lg, vv, acc, E);
    attn_finalize_kernel<<<(int)(((size_t)N * CDIM + 255) / 256), 256, 0, stream>>>(
        acc, ssum, sk, N);
    lnrelu(acc, idn, lnG, lnB, outh, CDIM);  // relu(LN(tconv + idn)) -> f16
  };

  // ---- input MLP ----
  gemm(xh, wtInp, inpB, pre, 64, 64);
  lnrelu(pre, nullptr, lninG, lninB, h64h, 64);

  // ---- transformer conv layer 1 (64 -> 256) ----
  attn_layer(h64h, 64, wtC1, c1B, wtRes1, res1B, ln1G, ln1B, h256h);

  // ---- transformer conv layer 2 (256 -> 256) ----
  attn_layer(h256h, CDIM, wtC2, c2B, wtRes2, res2B, ln2G, ln2B, h256h);

  // ---- output MLP ----
  gemm(h256h, wtOut1, out1B, pre, CDIM, 64);
  lnrelu(pre, nullptr, lnoG, lnoB, h64h, 64);
  gemm(h64h, wtOut2, out2B, (float*)d_out, 64, 32);
}
